// LinearInterpTrigram_65446711657254
// MI455X (gfx1250) — compile-verified, matrix operand order checked by reading, not run on hardware
//
#include <hip/hip_runtime.h>
#include <hip/hip_bf16.h>

// ---------------------------------------------------------------------------
// n-gram count accumulation (uni/bi/tri) for V=512, L=1e6 on MI455X (gfx1250).
//
// Roofline: dominated by the 538MB trigram table init (~46us @ 23.3TB/s);
// scatter adds are ~2M fp32 global atomics (~10us). Unigrams are privatized
// in LDS per workgroup; per-WG partials are reduced on the matrix unit with
// v_wmma_f32_16x16x32_f16 (all-ones A => column sums). Token staging uses the
// gfx1250 async global->LDS path (ASYNCcnt) when the builtin is available.
// ---------------------------------------------------------------------------

typedef __attribute__((ext_vector_type(4)))  float    f4;
typedef __attribute__((ext_vector_type(4)))  int      v4i;
typedef __attribute__((ext_vector_type(16))) _Float16 v16h;
typedef __attribute__((ext_vector_type(8)))  float    v8f;

#define NG_THREADS 256
#define NG_CHUNK   1024   // tokens per workgroup (4 per thread; <=2048 keeps f16 exact)
#define NG_MAXV    1024   // LDS histogram capacity

#if defined(__has_builtin)
#if __has_builtin(__builtin_amdgcn_global_load_async_to_lds_b128) && \
    __has_builtin(__builtin_amdgcn_s_wait_asynccnt)
#define USE_ASYNC_LDS 1
#else
#define USE_ASYNC_LDS 0
#endif
#else
#define USE_ASYNC_LDS 0
#endif

__device__ __forceinline__ void atomAddF(float* p, float v) {
#if defined(__HIP_PLATFORM_AMD__)
  unsafeAtomicAdd(p, v);   // native global_atomic_add_f32 (no CAS loop)
#else
  atomicAdd(p, v);
#endif
}

// --------------------------- 1) output init (copy) -------------------------
__global__ void copy_kernel(const float* __restrict__ src,
                            float* __restrict__ dst, size_t n) {
  size_t i = ((size_t)blockIdx.x * blockDim.x + threadIdx.x) * 4;
  const size_t stride = (size_t)gridDim.x * blockDim.x * 4;
  for (; i + 3 < n; i += stride) {
    f4 v = *(const f4*)(src + i);
    *(f4*)(dst + i) = v;
  }
  // ragged tail (at most one thread lands here; 4-aligned chunk starts)
  if (i < n) {
    for (; i < n; ++i) dst[i] = src[i];
  }
}

// --------------------------- 2) scatter counts -----------------------------
__global__ void __launch_bounds__(NG_THREADS)
ngram_count_kernel(const int* __restrict__ batch, int L, int V,
                   float* __restrict__ uni_out,
                   float* __restrict__ bi_out,
                   float* __restrict__ tri_out,
                   float* __restrict__ uni_partials,
                   int use_ws) {
  __shared__ int   toks[NG_CHUNK + 2];
  __shared__ float hist[NG_MAXV];

  const int tid  = threadIdx.x;
  const int base = blockIdx.x * NG_CHUNK;

  if (V <= NG_MAXV) {
    for (int b = tid; b < V; b += NG_THREADS) hist[b] = 0.0f;
  }

  const int g = base + tid * 4;
#if USE_ASYNC_LDS
  if (g + 3 < L) {
    // gfx1250 async copy: 16B per lane straight into LDS, tracked by ASYNCcnt.
    // Builtin expects (v4i AS1*, v4i AS3*, imm offset, imm cpol).
    v4i* gp = (v4i*)(batch + g);          // reinterpret (drops const)
    v4i* lp = (v4i*)(&toks[tid * 4]);     // generic pointer into LDS
    __builtin_amdgcn_global_load_async_to_lds_b128(
        (__attribute__((address_space(1))) v4i*)gp,
        (__attribute__((address_space(3))) v4i*)lp,
        /*offset=*/0, /*cpol=*/0);
  } else {
    for (int j = 0; j < 4; ++j)
      if (g + j < L) toks[tid * 4 + j] = batch[g + j];
  }
  if (tid < 2) {  // two boundary tokens for bigram/trigram lookahead
    const int e = base + NG_CHUNK + tid;
    if (e < L) toks[NG_CHUNK + tid] = batch[e];
  }
  __builtin_amdgcn_s_wait_asynccnt(0);
#else
  for (int j = 0; j < 4; ++j)
    if (g + j < L) toks[tid * 4 + j] = batch[g + j];
  if (tid < 2) {
    const int e = base + NG_CHUNK + tid;
    if (e < L) toks[NG_CHUNK + tid] = batch[e];
  }
#endif
  __syncthreads();

  for (int j = 0; j < 4; ++j) {
    const int li = tid * 4 + j;
    const int i  = base + li;
    if (i >= L) break;
    const int t0 = toks[li];
    if (V <= NG_MAXV) {
      atomicAdd(&hist[t0], 1.0f);                   // ds_add_f32, privatized
    } else {
      atomAddF(&uni_out[t0], 1.0f);                 // generic fallback
    }
    if (i + 1 < L) {
      const int t1 = toks[li + 1];
      atomAddF(&bi_out[(size_t)t0 * V + t1], 1.0f);
      if (i + 2 < L) {
        const int t2 = toks[li + 2];
        atomAddF(&tri_out[((size_t)t0 * V + (size_t)t1) * V + t2], 1.0f);
      }
    }
  }

  if (V <= NG_MAXV) {
    __syncthreads();
    for (int b = tid; b < V; b += NG_THREADS) {
      if (use_ws) {
        uni_partials[(size_t)blockIdx.x * V + b] = hist[b];  // full row written
      } else if (hist[b] != 0.0f) {
        atomAddF(&uni_out[b], hist[b]);
      }
    }
  }
}

// ------------------ 3) WMMA reduction of unigram partials ------------------
// D = A x B + C with A = all-ones 16x32 f16  =>  D[m,n] = sum_k B[k,n].
// Per-WG partial counts are <= NG_CHUNK (1024), exactly representable in f16.
// One wave per 16-column group; 32 partial rows per WMMA. Main loop does
// unconditional batched loads; the single tail step is branchless (cndmask
// on a clamped, always-valid address).
__global__ void __launch_bounds__(32)
uni_reduce_wmma_kernel(const float* __restrict__ partials, int nwg, int V,
                       const float* __restrict__ uni_in,
                       float* __restrict__ uni_out) {
  const int lane  = threadIdx.x;              // one wave32, EXEC all ones
  const int n     = blockIdx.x * 16 + (lane & 15);
  const int khalf = (lane < 16) ? 0 : 16;     // B 32x16: lanes 0-15 K=0..15, 16-31 K=16..31
  const size_t Vs = (size_t)V;

  v16h a;
#pragma unroll
  for (int j = 0; j < 16; ++j) a[j] = (_Float16)1.0f;
  v8f c = {};

  // Row pointer for this lane's half: row (kb + khalf), column n.
  const float* p = partials + (size_t)khalf * Vs + n;

  int kb = 0;
  for (; kb + 32 <= nwg; kb += 32) {
    float x[16];
    const float* q = p;
#pragma unroll
    for (int j = 0; j < 16; ++j) {   // 16 independent loads, one wait
      x[j] = *q;
      q += Vs;
    }
    v16h b;
#pragma unroll
    for (int j = 0; j < 16; ++j) b[j] = (_Float16)x[j];
    c = __builtin_amdgcn_wmma_f32_16x16x32_f16(
        false, a, false, b, (short)0, c, false, false);
    p += 32 * Vs;
  }

  if (kb < nwg) {  // tail: branchless masking, clamped addresses stay in-bounds
    v16h b;
#pragma unroll
    for (int j = 0; j < 16; ++j) {
      const int kk = kb + khalf + j;
      const int kc = (kk < nwg) ? kk : (nwg - 1);
      const float x = partials[(size_t)kc * Vs + n];
      b[j] = (_Float16)((kk < nwg) ? x : 0.0f);
    }
    c = __builtin_amdgcn_wmma_f32_16x16x32_f16(
        false, a, false, b, (short)0, c, false, false);
  }

  // C/D layout: VGPR0 holds M=0 row, N = lane for lanes 0-15.
  if (lane < 16) uni_out[n] = uni_in[n] + c[0];
}

// ------------------------------- launcher ----------------------------------
extern "C" void kernel_launch(void* const* d_in, const int* in_sizes, int n_in,
                              void* d_out, int out_size, void* d_ws,
                              size_t ws_size, hipStream_t stream) {
  const float* uni_in = (const float*)d_in[0];
  const float* bi_in  = (const float*)d_in[1];
  const float* tri_in = (const float*)d_in[2];
  const int*   batch  = (const int*)d_in[3];

  const int V = in_sizes[0];
  const int L = in_sizes[3];
  const size_t nuni = (size_t)in_sizes[0];
  const size_t nbi  = (size_t)in_sizes[1];
  const size_t ntri = (size_t)in_sizes[2];

  float* out     = (float*)d_out;
  float* uni_out = out;
  float* bi_out  = out + nuni;
  float* tri_out = out + nuni + nbi;

  auto copy_blocks = [](size_t n) -> int {
    size_t b = (n / 4 + 255) / 256;
    if (b < 1) b = 1;
    if (b > 4096) b = 4096;
    return (int)b;
  };

  // 1) out = in (also clears the 0xAA poison); b128 streaming copies
  copy_kernel<<<copy_blocks(nuni), 256, 0, stream>>>(uni_in, uni_out, nuni);
  copy_kernel<<<copy_blocks(nbi),  256, 0, stream>>>(bi_in,  bi_out,  nbi);
  copy_kernel<<<copy_blocks(ntri), 256, 0, stream>>>(tri_in, tri_out, ntri);

  // 2) scatter counts (same-stream ordering after the copies)
  const int  nblk   = (L + NG_CHUNK - 1) / NG_CHUNK;
  const bool can_ws = (V <= NG_MAXV) && (V % 16 == 0) && (d_ws != nullptr) &&
                      (ws_size >= (size_t)nblk * (size_t)V * sizeof(float));
  float* partials = (float*)d_ws;

  ngram_count_kernel<<<nblk, NG_THREADS, 0, stream>>>(
      batch, L, V, uni_out, bi_out, tri_out, partials, can_ws ? 1 : 0);

  // 3) matrix-unit reduction of per-WG unigram histograms
  if (can_ws) {
    uni_reduce_wmma_kernel<<<V / 16, 32, 0, stream>>>(partials, nblk, V,
                                                      uni_in, uni_out);
  }
}
